// MultiheadAttention_76544907149528
// MI455X (gfx1250) — compile-verified, matrix operand order checked by reading, not run on hardware
//
#include <hip/hip_runtime.h>
#include <hip/hip_bf16.h>
#include <math.h>
#include <cstddef>

// ---------------------------------------------------------------------------
// MI455X / gfx1250 multihead attention, bf16 WMMA (16x16x32) with f32 accum.
// B=2, T=S=2048, C=1024, H=16, D=64.
// ---------------------------------------------------------------------------

typedef __attribute__((ext_vector_type(16))) __bf16 bf16x16_t;
typedef __attribute__((ext_vector_type(8)))  __bf16 bf16x8_t;
typedef __attribute__((ext_vector_type(8)))  float  f32x8_t;

#define WMMA_BF16(A, Bm, Cm) \
  __builtin_amdgcn_wmma_f32_16x16x32_bf16(false, (A), false, (Bm), (short)0, (Cm), false, false)

static constexpr int Bb = 2, Hh = 16, Tt = 2048, Ss = 2048, Dd = 64, Cc = 1024;

__device__ __forceinline__ bf16x16_t make_frag(const __bf16* p0, const __bf16* p1) {
  bf16x8_t lo = *(const bf16x8_t*)p0;   // 16-byte load (b128)
  bf16x8_t hi = *(const bf16x8_t*)p1;   // 16-byte load (b128)
  return __builtin_shufflevector(lo, hi, 0, 1, 2, 3, 4, 5, 6, 7, 8, 9, 10, 11, 12, 13, 14, 15);
}

__device__ __forceinline__ f32x8_t zero8() {
  return (f32x8_t){0.f, 0.f, 0.f, 0.f, 0.f, 0.f, 0.f, 0.f};
}

// ---------------------------------------------------------------------------
// f32 -> bf16 elementwise convert (bandwidth trivial vs 23.3 TB/s HBM)
// ---------------------------------------------------------------------------
__global__ void cvt_f32_to_bf16(const float* __restrict__ x, __bf16* __restrict__ y, int n) {
  int i = blockIdx.x * blockDim.x + threadIdx.x;
  int stride = gridDim.x * blockDim.x;
  for (; i < n; i += stride) y[i] = (__bf16)x[i];
}

// ---------------------------------------------------------------------------
// Y = X @ W^T + bias.  X:[M,K] bf16 row-major, W:[N,K] bf16 row-major.
// One wave computes a 64(M) x 64(N) tile: 4 A-frags x 4 B-frags -> 16 WMMAs
// per K-step of 32.  Explicit ping-pong pipeline (unrolled by 2): while the
// 16 WMMAs of one K-step run, the 16 b128 loads of the next are in flight
// (LOADcnt is 6 bits -> 32 outstanding loads is fine).  No buffer copies.
// mode 0: bf16 out, layout [B,H,L,64]   (Q with scale, K)
// mode 1: bf16 out, layout [B,H,64,L]   (V transposed for PV B-operand)
// mode 2: f32  out, row-major [M,N]     (final projection -> d_out)
// ---------------------------------------------------------------------------
__device__ __forceinline__ void load_frags(const __bf16* const* Arow,
                                           const __bf16* const* Brow,
                                           int ks, int k0,
                                           bf16x16_t a[4], bf16x16_t bfr[4]) {
#pragma unroll
  for (int mi = 0; mi < 4; ++mi)
    a[mi] = make_frag(Arow[mi] + ks + k0, Arow[mi] + ks + k0 + 16);
#pragma unroll
  for (int j = 0; j < 4; ++j)
    bfr[j] = make_frag(Brow[j] + ks + k0, Brow[j] + ks + k0 + 16);
}

__device__ __forceinline__ void wmma_step(const bf16x16_t a[4], const bf16x16_t bfr[4],
                                          f32x8_t acc[4][4]) {
#pragma unroll
  for (int j = 0; j < 4; ++j)
#pragma unroll
    for (int mi = 0; mi < 4; ++mi)
      acc[mi][j] = WMMA_BF16(a[mi], bfr[j], acc[mi][j]);
}

__global__ __launch_bounds__(32) void linear_wmma(
    const __bf16* __restrict__ X, const __bf16* __restrict__ W,
    const float* __restrict__ bias, void* __restrict__ Y,
    int M, int N, int K, int L, int mode, float out_scale) {
  const int lane = threadIdx.x;
  const int lrow = lane & 15;
  const int k0   = (lane >> 4) * 8;   // A/B fragment K phase per half-wave
  const int tm = blockIdx.x, tn = blockIdx.y;

  const __bf16* Arow[4];
#pragma unroll
  for (int mi = 0; mi < 4; ++mi)
    Arow[mi] = X + (size_t)(tm * 64 + mi * 16 + lrow) * K;
  const __bf16* Brow[4];
#pragma unroll
  for (int j = 0; j < 4; ++j)
    Brow[j] = W + (size_t)(tn * 64 + j * 16 + lrow) * K;

  float bj[4];
#pragma unroll
  for (int j = 0; j < 4; ++j) bj[j] = bias[tn * 64 + j * 16 + lrow];

  f32x8_t acc[4][4];
#pragma unroll
  for (int mi = 0; mi < 4; ++mi)
#pragma unroll
    for (int j = 0; j < 4; ++j) acc[mi][j] = zero8();

  // ---- ping-pong fragment pipeline (K assumed a multiple of 64) -----------
  bf16x16_t a0[4], b0[4], a1[4], b1[4];
  load_frags(Arow, Brow, 0, k0, a0, b0);
  for (int ks = 0; ks < K; ks += 64) {
    // even half: compute ks with buf0 while loading ks+32 into buf1
    load_frags(Arow, Brow, ks + 32, k0, a1, b1);
    __builtin_prefetch((const void*)(Arow[0] + ks + 64), 0, 0);  // global_prefetch_b8
    wmma_step(a0, b0, acc);
    // odd half: compute ks+32 with buf1 while loading ks+64 into buf0
    if (ks + 64 < K) load_frags(Arow, Brow, ks + 64, k0, a0, b0);
    wmma_step(a1, b1, acc);
  }

  const int hi8 = (lane >> 4) * 8;   // C-layout: VGPR r holds row r (+8 for lanes 16..31)
#pragma unroll
  for (int mi = 0; mi < 4; ++mi) {
#pragma unroll
    for (int j = 0; j < 4; ++j) {
#pragma unroll
      for (int r = 0; r < 8; ++r) {
        float v = (acc[mi][j][r] + bj[j]) * out_scale;
        int m = tm * 64 + mi * 16 + r + hi8;
        int n = tn * 64 + j * 16 + lrow;
        if (mode == 2) {
          ((float*)Y)[(size_t)m * N + n] = v;
        } else {
          int bb = m / L, t = m % L;
          int h = n >> 6, d = n & 63;
          size_t idx = (mode == 0)
                           ? ((size_t)(bb * Hh + h) * L + t) * 64 + d
                           : ((size_t)(bb * Hh + h) * 64 + d) * L + t;
          ((__bf16*)Y)[idx] = (__bf16)v;
        }
      }
    }
  }
}

// ---------------------------------------------------------------------------
// Flash attention: one wave per (b, h, 32-row Q tile).  Two 16-row Q sub-tiles
// share every K-fragment and every V-fragment (1 b128 load : 1 WMMA).
// V-fragment loads are issued BEFORE the softmax VALU block so their latency
// hides behind ~100 VALU ops (shuffle reductions + exp2).
// Qh: [B,H,T,64] bf16 (already scaled by 1/sqrt(D) via linear out_scale)
// Kh: [B,H,S,64] bf16
// Vt: [B,H,64,S] bf16  (transposed so PV B-fragments are contiguous)
// Oo: [B,T,C]   bf16   (heads re-interleaved for the output projection)
// ---------------------------------------------------------------------------
__global__ __launch_bounds__(32) void attn_flash(
    const __bf16* __restrict__ Qh, const __bf16* __restrict__ Kh,
    const __bf16* __restrict__ Vt, __bf16* __restrict__ Oo) {
  __shared__ __align__(16) __bf16 Plds[32 * 40];  // 32x32 P tile, 80B row stride

  const int lane = threadIdx.x;
  const int lrow = lane & 15;
  const int hi   = lane >> 4;
  const int k0   = hi * 8;

  const int qt = blockIdx.x % (Tt / 32);
  const int bh = blockIdx.x / (Tt / 32);
  const int h  = bh % Hh;
  const int b  = bh / Hh;

  const __bf16* Qp = Qh + ((size_t)bh * Tt + qt * 32) * Dd;
  const __bf16* Kp = Kh + (size_t)bh * Ss * Dd;
  const __bf16* Vp = Vt + (size_t)bh * Dd * Ss;

  // Q A-fragments: 2 sub-tiles x 2 K-steps of the D=64 contraction
  bf16x16_t aq[2][2];
#pragma unroll
  for (int qi = 0; qi < 2; ++qi)
#pragma unroll
    for (int kk = 0; kk < 2; ++kk)
      aq[qi][kk] = make_frag(Qp + (qi * 16 + lrow) * Dd + kk * 32 + k0,
                             Qp + (qi * 16 + lrow) * Dd + kk * 32 + k0 + 16);

  float m_run[2][8], l_run[2][8];
#pragma unroll
  for (int qi = 0; qi < 2; ++qi)
#pragma unroll
    for (int r = 0; r < 8; ++r) { m_run[qi][r] = -1e30f; l_run[qi][r] = 0.f; }
  f32x8_t o[2][4];
#pragma unroll
  for (int qi = 0; qi < 2; ++qi)
#pragma unroll
    for (int jd = 0; jd < 4; ++jd) o[qi][jd] = zero8();

  // exp2 domain: fold log2(e) into the score scale (1/sqrt(D) already in Q)
  const float kScale = 1.44269504088896340736f;

  for (int sb = 0; sb < Ss; sb += 32) {
    // ---- batched K-fragment loads (one clause of 8 b128s) ------------------
    bf16x16_t kb[2][2];   // [kk][n-subtile]
#pragma unroll
    for (int kk = 0; kk < 2; ++kk) {
      kb[kk][0] = make_frag(Kp + (size_t)(sb + lrow) * Dd + kk * 32 + k0,
                            Kp + (size_t)(sb + lrow) * Dd + kk * 32 + k0 + 16);
      kb[kk][1] = make_frag(Kp + (size_t)(sb + 16 + lrow) * Dd + kk * 32 + k0,
                            Kp + (size_t)(sb + 16 + lrow) * Dd + kk * 32 + k0 + 16);
    }
    // ---- V-fragment loads issued early; consumed after the softmax ---------
    bf16x16_t vb[4];
#pragma unroll
    for (int jd = 0; jd < 4; ++jd) {
      const __bf16* vp = Vp + (size_t)(jd * 16 + lrow) * Ss + sb;
      vb[jd] = make_frag(vp + k0, vp + k0 + 16);
    }
    if (sb + 32 < Ss)
      __builtin_prefetch((const void*)(Kp + (size_t)(sb + 32 + lrow) * Dd), 0, 0);

    // ---- scores: 32 x 32 tile of Q K^T (8 WMMAs) ---------------------------
    f32x8_t s[2][2];
#pragma unroll
    for (int qi = 0; qi < 2; ++qi) { s[qi][0] = zero8(); s[qi][1] = zero8(); }
#pragma unroll
    for (int kk = 0; kk < 2; ++kk)
#pragma unroll
      for (int qi = 0; qi < 2; ++qi) {
        s[qi][0] = WMMA_BF16(aq[qi][kk], kb[kk][0], s[qi][0]);
        s[qi][1] = WMMA_BF16(aq[qi][kk], kb[kk][1], s[qi][1]);
      }

    // ---- online softmax in C layout: row r lives in VGPR r of a 16-lane half
#pragma unroll
    for (int qi = 0; qi < 2; ++qi) {
      float corr[8];
#pragma unroll
      for (int r = 0; r < 8; ++r) {
        float v0 = s[qi][0][r] * kScale;
        float v1 = s[qi][1][r] * kScale;
        float mb = fmaxf(v0, v1);
        mb = fmaxf(mb, __shfl_xor(mb, 1));
        mb = fmaxf(mb, __shfl_xor(mb, 2));
        mb = fmaxf(mb, __shfl_xor(mb, 4));
        mb = fmaxf(mb, __shfl_xor(mb, 8));   // reduced within each 16-lane half
        float mn = fmaxf(m_run[qi][r], mb);
        corr[r] = exp2f(m_run[qi][r] - mn);
        m_run[qi][r] = mn;
        float e0 = exp2f(v0 - mn);
        float e1 = exp2f(v1 - mn);
        s[qi][0][r] = e0;
        s[qi][1][r] = e1;
        float ps = e0 + e1;
        ps += __shfl_xor(ps, 1);
        ps += __shfl_xor(ps, 2);
        ps += __shfl_xor(ps, 4);
        ps += __shfl_xor(ps, 8);
        l_run[qi][r] = l_run[qi][r] * corr[r] + ps;
      }
      f32x8_t corrv = {corr[0], corr[1], corr[2], corr[3],
                       corr[4], corr[5], corr[6], corr[7]};
#pragma unroll
      for (int jd = 0; jd < 4; ++jd) o[qi][jd] *= corrv;

      // C-layout P -> LDS (row-major 32x32 bf16)
#pragma unroll
      for (int r = 0; r < 8; ++r) {
        int row = qi * 16 + r + hi * 8;
        Plds[row * 40 + lrow]      = (__bf16)s[qi][0][r];
        Plds[row * 40 + 16 + lrow] = (__bf16)s[qi][1][r];
      }
    }
    __syncthreads();
    bf16x16_t pa0 = make_frag(&Plds[lrow * 40 + k0],
                              &Plds[lrow * 40 + k0 + 16]);
    bf16x16_t pa1 = make_frag(&Plds[(16 + lrow) * 40 + k0],
                              &Plds[(16 + lrow) * 40 + k0 + 16]);
    __syncthreads();

    // ---- O += P V : each preloaded V-fragment feeds both Q sub-tiles -------
#pragma unroll
    for (int jd = 0; jd < 4; ++jd) {
      o[0][jd] = WMMA_BF16(pa0, vb[jd], o[0][jd]);
      o[1][jd] = WMMA_BF16(pa1, vb[jd], o[1][jd]);
    }
  }

  // ---- normalize and store as [B,T,C] with head-major columns -------------
#pragma unroll
  for (int qi = 0; qi < 2; ++qi) {
#pragma unroll
    for (int r = 0; r < 8; ++r) {
      float inv = 1.f / l_run[qi][r];
      int t = qt * 32 + qi * 16 + r + hi * 8;
#pragma unroll
      for (int jd = 0; jd < 4; ++jd) {
        int col = h * 64 + jd * 16 + lrow;
        Oo[((size_t)(b * Tt + t)) * Cc + col] = (__bf16)(o[qi][jd][r] * inv);
      }
    }
  }
}

// ---------------------------------------------------------------------------
// Host-side orchestration
// ---------------------------------------------------------------------------
extern "C" void kernel_launch(void* const* d_in, const int* in_sizes, int n_in,
                              void* d_out, int out_size, void* d_ws, size_t ws_size,
                              hipStream_t stream) {
  (void)in_sizes; (void)n_in; (void)out_size;

  const float* q  = (const float*)d_in[0];
  const float* k  = (const float*)d_in[1];
  const float* v  = (const float*)d_in[2];
  const float* Wq = (const float*)d_in[3];
  const float* bq = (const float*)d_in[4];
  const float* Wk = (const float*)d_in[5];
  const float* bk = (const float*)d_in[6];
  const float* Wv = (const float*)d_in[7];
  const float* bv = (const float*)d_in[8];
  const float* Wo = (const float*)d_in[9];
  const float* bo = (const float*)d_in[10];

  const int Mtok = Bb * Tt;          // 4096
  const int nX   = Mtok * Cc;        // 4,194,304
  const int nW   = Cc * Cc;          // 1,048,576

  // Workspace carve-up (all chunks 256B-aligned multiples): 64 MiB total.
  char* ws = (char*)d_ws;
  size_t off = 0;
  auto take = [&](size_t bytes) {
    size_t p = off;
    off += (bytes + 255) & ~(size_t)255;
    return p;
  };
  __bf16* Xq  = (__bf16*)(ws + take((size_t)nX * 2));
  __bf16* Xk  = (__bf16*)(ws + take((size_t)nX * 2));
  __bf16* Xv  = (__bf16*)(ws + take((size_t)nX * 2));
  __bf16* Wqb = (__bf16*)(ws + take((size_t)nW * 2));
  __bf16* Wkb = (__bf16*)(ws + take((size_t)nW * 2));
  __bf16* Wvb = (__bf16*)(ws + take((size_t)nW * 2));
  __bf16* Wob = (__bf16*)(ws + take((size_t)nW * 2));
  __bf16* Qh  = (__bf16*)(ws + take((size_t)nX * 2));
  __bf16* Kh  = (__bf16*)(ws + take((size_t)nX * 2));
  __bf16* Vt  = (__bf16*)(ws + take((size_t)nX * 2));
  __bf16* Ao  = (__bf16*)(ws + take((size_t)nX * 2));
  if (off > ws_size) return;  // workspace too small; bail deterministically

  // 1) fp32 -> bf16 converts
  dim3 cb(256), cg(512);
  hipLaunchKernelGGL(cvt_f32_to_bf16, cg, cb, 0, stream, q,  Xq,  nX);
  hipLaunchKernelGGL(cvt_f32_to_bf16, cg, cb, 0, stream, k,  Xk,  nX);
  hipLaunchKernelGGL(cvt_f32_to_bf16, cg, cb, 0, stream, v,  Xv,  nX);
  hipLaunchKernelGGL(cvt_f32_to_bf16, cg, cb, 0, stream, Wq, Wqb, nW);
  hipLaunchKernelGGL(cvt_f32_to_bf16, cg, cb, 0, stream, Wk, Wkb, nW);
  hipLaunchKernelGGL(cvt_f32_to_bf16, cg, cb, 0, stream, Wv, Wvb, nW);
  hipLaunchKernelGGL(cvt_f32_to_bf16, cg, cb, 0, stream, Wo, Wob, nW);

  // 2) projections (64x64 tiles per wave, ping-pong K pipeline)
  dim3 gg(Mtok / 64, Cc / 64), gb(32);
  const float qscale = 0.125f;  // 1/sqrt(D), fused into Q
  hipLaunchKernelGGL(linear_wmma, gg, gb, 0, stream, Xq, Wqb, bq, (void*)Qh,
                     Mtok, Cc, Cc, Tt, 0, qscale);
  hipLaunchKernelGGL(linear_wmma, gg, gb, 0, stream, Xk, Wkb, bk, (void*)Kh,
                     Mtok, Cc, Cc, Ss, 0, 1.0f);
  hipLaunchKernelGGL(linear_wmma, gg, gb, 0, stream, Xv, Wvb, bv, (void*)Vt,
                     Mtok, Cc, Cc, Ss, 1, 1.0f);

  // 3) flash attention: one wave per 32-row Q tile per (b,h)
  hipLaunchKernelGGL(attn_flash, dim3(Bb * Hh * (Tt / 32)), dim3(32), 0, stream,
                     Qh, Kh, Vt, Ao);

  // 4) output projection -> f32 d_out
  hipLaunchKernelGGL(linear_wmma, gg, gb, 0, stream, Ao, Wob, bo, d_out,
                     Mtok, Cc, Cc, Tt, 2, 1.0f);
}
